// MemoryModule_88287347737201
// MI455X (gfx1250) — matrix operand order repaired and checked by reading, not compiled, not run on hardware
//
#include <hip/hip_runtime.h>
#include <hip/hip_bf16.h>

typedef _Float16 h16;
typedef __attribute__((ext_vector_type(16))) _Float16 v16h;
typedef __attribute__((ext_vector_type(8)))  float    v8f;
typedef __attribute__((ext_vector_type(4)))  float    v4f;

#define HDIM 128
#define EFD  64
#define CW   320          // combined width = 2H + EF
#define LSTR 136          // LDS row stride in halves (128 + 8 pad -> bank spread)
#define TSZ  (16 * LSTR)  // one 16-row f16 tile buffer (2176 halves = 4352 B)

// workspace layout (half-element offsets)
#define OFF_W1   0                      // W1 f16:              128 x 320
#define OFF_WF   40960                  // (W_ih @ W2) f16:     384 x 128
#define OFF_WHH  90112                  // W_hh f16:            384 x 128
#define OFF_BF   139264                 // b_fused f32[384] = W_ih@b2 + b_ih

__device__ __forceinline__ v8f wzero() {
  v8f z;
#pragma unroll
  for (int i = 0; i < 8; ++i) z[i] = 0.0f;
  return z;
}

__device__ __forceinline__ v8f wmma16(v16h a, v16h b, v8f c) {
  // D = A(16x32 f16) x B(32x16 f16) + C(16x16 f32)
  return __builtin_amdgcn_wmma_f32_16x16x32_f16(false, a, false, b, (short)0, c,
                                                false, false);
}

// B fragment: lane (c,g) holds column N=c, halves 0..7 = K 8g.., halves 8..15 =
// K 16+8g.. (two 16B chunks) when B is stored K-contiguous per output row (W row-major).
__device__ __forceinline__ v16h load_bfrag(const h16* p) {
  union { v16h v; uint4 q[2]; } u;
  const uint4* s = (const uint4*)p;
  u.q[0] = s[0];
  u.q[1] = s[2];
  return u.v;
}

// A fragment from two 8-float contiguous chunks (global f32 -> f16 cvt).
// Lane (r,g): halves 0..7 = K kb+8g.., halves 8..15 = K kb+16+8g..
__device__ __forceinline__ v16h cvt_afrag(const float* c0, const float* c1) {
  const v4f* p0 = (const v4f*)c0;
  const v4f* p1 = (const v4f*)c1;
  v4f x0 = p0[0], x1 = p0[1], y0 = p1[0], y1 = p1[1];
  v16h a;
#pragma unroll
  for (int i = 0; i < 4; ++i) {
    a[i]      = (h16)x0[i];
    a[4 + i]  = (h16)x1[i];
    a[8 + i]  = (h16)y0[i];
    a[12 + i] = (h16)y1[i];
  }
  return a;
}

// A fragment rebuilt from an LDS-staged f16 row-major [16][LSTR] tile.
__device__ __forceinline__ v16h lds_afrag(const h16* stage, int r, int g, int kt) {
  union { v16h v; uint4 q[2]; } u;
  u.q[0] = *(const uint4*)(stage + r * LSTR + kt * 32 + 8 * g);
  u.q[1] = *(const uint4*)(stage + r * LSTR + kt * 32 + 16 + 8 * g);
  return u.v;
}

// Inverse of lds_afrag: scatter a fragment's halves to row-major LDS.
__device__ __forceinline__ void store_afrag(h16* stage, int r, int g, int kt,
                                            v16h a) {
  union { v16h v; uint4 q[2]; } u;
  u.v = a;
  *(uint4*)(stage + r * LSTR + kt * 32 + 8 * g)      = u.q[0];
  *(uint4*)(stage + r * LSTR + kt * 32 + 16 + 8 * g) = u.q[1];
}

__device__ __forceinline__ float sigmoidf(float x) {
  return 1.0f / (1.0f + __expf(-x));
}

// W1, W_hh f32 -> f16
__global__ void cvt_weights_kernel(const float* __restrict__ W1,
                                   const float* __restrict__ Whh,
                                   h16* __restrict__ ws) {
  int i = blockIdx.x * blockDim.x + threadIdx.x;
  if (i < 40960) ws[OFF_W1 + i] = (h16)W1[i];
  if (i < 49152) ws[OFF_WHH + i] = (h16)Whh[i];
}

// W_fused = W_ih @ W2 (f32 accumulate, cast f16); b_fused = W_ih @ b2 + b_ih
__global__ void fuse_weights_kernel(const float* __restrict__ Wih,
                                    const float* __restrict__ W2,
                                    const float* __restrict__ b2,
                                    const float* __restrict__ b_ih,
                                    h16* __restrict__ ws) {
  int i = blockIdx.x * blockDim.x + threadIdx.x;   // 49152 threads
  if (i >= 49152) return;
  const int a = i >> 7;       // 0..383
  const int c = i & 127;      // 0..127
  float s = 0.0f;
  for (int t = 0; t < 128; ++t) s += Wih[a * 128 + t] * W2[t * 128 + c];
  ws[OFF_WF + a * 128 + c] = (h16)s;
  if (c == 0) {
    float sb = 0.0f;
    for (int t = 0; t < 128; ++t) sb += Wih[a * 128 + t] * b2[t];
    ((float*)(ws + OFF_BF))[a] = sb + b_ih[a];
  }
}

__global__ __launch_bounds__(256) void
tgn_update_kernel(const float* __restrict__ memory,
                  const int*   __restrict__ src_ids,
                  const int*   __restrict__ dst_ids,
                  const float* __restrict__ edge_feats,
                  const h16*   __restrict__ ws,
                  const float* __restrict__ b1,
                  const float* __restrict__ b_hh,
                  float* __restrict__ out,
                  int E) {
  // Per wave: 4 tile buffers (src0, src1, hid0, hid1) = 17408 B; block: 139264 B
  __shared__ h16 ldsAll[8 * 4 * TSZ];

  const int tid  = threadIdx.x;
  const int wave = tid >> 5;
  const int lane = tid & 31;
  const int r    = lane & 15;   // A row / B column / C column
  const int g    = lane >> 4;

  const int tile = blockIdx.x * 8 + wave;   // one wave = 32 edges (2 M-tiles)
  const int m0   = tile * 32;
  if (m0 >= E) return;          // uniform per wave -> EXEC all-ones inside

  h16* Ssrc0 = ldsAll + wave * 4 * TSZ;
  h16* Ssrc1 = Ssrc0 + TSZ;
  h16* Shid0 = Ssrc0 + 2 * TSZ;
  h16* Shid1 = Ssrc0 + 3 * TSZ;

  const h16*   W1h = ws + OFF_W1;
  const h16*   Wfh = ws + OFF_WF;
  const h16*   Whh = ws + OFF_WHH;
  const float* bf  = (const float*)(ws + OFF_BF);

  const long src0 = (long)src_ids[m0 + r];
  const long dst0 = (long)dst_ids[m0 + r];
  const long src1 = (long)src_ids[m0 + 16 + r];
  const long dst1 = (long)dst_ids[m0 + 16 + r];

  const float* srcp0 = memory + src0 * HDIM;
  const float* dstp0 = memory + dst0 * HDIM;
  const float* efp0  = edge_feats + (long)(m0 + r) * EFD;
  const float* srcp1 = memory + src1 * HDIM;
  const float* dstp1 = memory + dst1 * HDIM;
  const float* efp1  = edge_feats + (long)(m0 + 16 + r) * EFD;

  // ------------- GEMM1: hid = relu(combined @ W1.T + b1), N split in halves
#pragma unroll
  for (int h = 0; h < 2; ++h) {
    v8f acc0[4], acc1[4];
#pragma unroll
    for (int nt = 0; nt < 4; ++nt) { acc0[nt] = wzero(); acc1[nt] = wzero(); }

    // Region 1: K 0..127 = src rows (staged to LDS once, for the gh GEMM)
#pragma unroll
    for (int kt = 0; kt < 4; ++kt) {
      v16h a0 = cvt_afrag(srcp0 + kt * 32 + 8 * g, srcp0 + kt * 32 + 16 + 8 * g);
      v16h a1 = cvt_afrag(srcp1 + kt * 32 + 8 * g, srcp1 + kt * 32 + 16 + 8 * g);
      if (h == 0) {
        store_afrag(Ssrc0, r, g, kt, a0);
        store_afrag(Ssrc1, r, g, kt, a1);
      }
#pragma unroll
      for (int nt = 0; nt < 4; ++nt) {
        v16h b = load_bfrag(W1h + ((h * 4 + nt) * 16 + r) * CW + kt * 32 + 8 * g);
        acc0[nt] = wmma16(a0, b, acc0[nt]);
        acc1[nt] = wmma16(a1, b, acc1[nt]);
      }
    }
    // Region 2: K 128..255 = dst rows
#pragma unroll
    for (int kt = 0; kt < 4; ++kt) {
      v16h a0 = cvt_afrag(dstp0 + kt * 32 + 8 * g, dstp0 + kt * 32 + 16 + 8 * g);
      v16h a1 = cvt_afrag(dstp1 + kt * 32 + 8 * g, dstp1 + kt * 32 + 16 + 8 * g);
#pragma unroll
      for (int nt = 0; nt < 4; ++nt) {
        v16h b = load_bfrag(W1h + ((h * 4 + nt) * 16 + r) * CW + 128 + kt * 32 + 8 * g);
        acc0[nt] = wmma16(a0, b, acc0[nt]);
        acc1[nt] = wmma16(a1, b, acc1[nt]);
      }
    }
    // Region 3: K 256..319 = edge features
#pragma unroll
    for (int kt = 0; kt < 2; ++kt) {
      v16h a0 = cvt_afrag(efp0 + kt * 32 + 8 * g, efp0 + kt * 32 + 16 + 8 * g);
      v16h a1 = cvt_afrag(efp1 + kt * 32 + 8 * g, efp1 + kt * 32 + 16 + 8 * g);
#pragma unroll
      for (int nt = 0; nt < 4; ++nt) {
        v16h b = load_bfrag(W1h + ((h * 4 + nt) * 16 + r) * CW + 256 + kt * 32 + 8 * g);
        acc0[nt] = wmma16(a0, b, acc0[nt]);
        acc1[nt] = wmma16(a1, b, acc1[nt]);
      }
    }
    // bias + relu, stage hid columns of this half
#pragma unroll
    for (int nt = 0; nt < 4; ++nt) {
      const float bias = b1[(h * 4 + nt) * 16 + r];
#pragma unroll
      for (int p = 0; p < 8; ++p) {
        float h0 = acc0[nt][p] + bias;
        float h1 = acc1[nt][p] + bias;
        Shid0[(p + 8 * g) * LSTR + (h * 4 + nt) * 16 + r] =
            (h16)(h0 > 0.0f ? h0 : 0.0f);
        Shid1[(p + 8 * g) * LSTR + (h * 4 + nt) * 16 + r] =
            (h16)(h1 > 0.0f ? h1 : 0.0f);
      }
    }
  }
  asm volatile("s_wait_dscnt 0x0" ::: "memory");

  int sid0[8], sid1[8];
#pragma unroll
  for (int p = 0; p < 8; ++p) {
    sid0[p] = src_ids[m0 + p + 8 * g];
    sid1[p] = src_ids[m0 + 16 + p + 8 * g];
  }

  // ------------- GRU: gi = hid @ Wf.T + bf ; gh = src @ W_hh.T + b_hh ------
  // r,z gates need only gi+gh  -> one shared accumulator per gate
  // n gate needs gi_n, gh_n separately (n = tanh(gi_n + r * gh_n))
#pragma unroll 1
  for (int j = 0; j < 8; ++j) {         // 16-wide output column tiles
    v8f rS0 = wzero(), zS0 = wzero(), iN0 = wzero(), hN0 = wzero();
    v8f rS1 = wzero(), zS1 = wzero(), iN1 = wzero(), hN1 = wzero();
#pragma unroll
    for (int kt = 0; kt < 4; ++kt) {
      const int ko = kt * 32 + 8 * g;
      v16h aM0 = lds_afrag(Shid0, r, g, kt);
      v16h aM1 = lds_afrag(Shid1, r, g, kt);
      v16h aS0 = lds_afrag(Ssrc0, r, g, kt);
      v16h aS1 = lds_afrag(Ssrc1, r, g, kt);
      v16h bR = load_bfrag(Wfh + (           j * 16 + r) * HDIM + ko);
      v16h bZ = load_bfrag(Wfh + (HDIM     + j * 16 + r) * HDIM + ko);
      v16h bN = load_bfrag(Wfh + (2 * HDIM + j * 16 + r) * HDIM + ko);
      v16h cR = load_bfrag(Whh + (           j * 16 + r) * HDIM + ko);
      v16h cZ = load_bfrag(Whh + (HDIM     + j * 16 + r) * HDIM + ko);
      v16h cN = load_bfrag(Whh + (2 * HDIM + j * 16 + r) * HDIM + ko);
      rS0 = wmma16(aM0, bR, rS0);  rS0 = wmma16(aS0, cR, rS0);
      zS0 = wmma16(aM0, bZ, zS0);  zS0 = wmma16(aS0, cZ, zS0);
      iN0 = wmma16(aM0, bN, iN0);  hN0 = wmma16(aS0, cN, hN0);
      rS1 = wmma16(aM1, bR, rS1);  rS1 = wmma16(aS1, cR, rS1);
      zS1 = wmma16(aM1, bZ, zS1);  zS1 = wmma16(aS1, cZ, zS1);
      iN1 = wmma16(aM1, bN, iN1);  hN1 = wmma16(aS1, cN, hN1);
    }
    const int col = j * 16 + r;
    const float bRt = bf[col]            + b_hh[col];
    const float bZt = bf[HDIM + col]     + b_hh[HDIM + col];
    const float bNi = bf[2 * HDIM + col];
    const float bNh = b_hh[2 * HDIM + col];
#pragma unroll
    for (int p = 0; p < 8; ++p) {
      const long s = (long)sid0[p];
      const float hv = memory[s * HDIM + col];
      const float rr = sigmoidf(rS0[p] + bRt);
      const float zz = sigmoidf(zS0[p] + bZt);
      const float nn = tanhf(iN0[p] + bNi + rr * (hN0[p] + bNh));
      out[s * HDIM + col] = (1.0f - zz) * nn + zz * hv;
    }
#pragma unroll
    for (int p = 0; p < 8; ++p) {
      const long s = (long)sid1[p];
      const float hv = memory[s * HDIM + col];
      const float rr = sigmoidf(rS1[p] + bRt);
      const float zz = sigmoidf(zS1[p] + bZt);
      const float nn = tanhf(iN1[p] + bNi + rr * (hN1[p] + bNh));
      out[s * HDIM + col] = (1.0f - zz) * nn + zz * hv;
    }
  }
}

extern "C" void kernel_launch(void* const* d_in, const int* in_sizes, int n_in,
                              void* d_out, int out_size, void* d_ws, size_t ws_size,
                              hipStream_t stream) {
  const float* memory     = (const float*)d_in[0];
  const int*   src_ids    = (const int*)  d_in[1];
  const int*   dst_ids    = (const int*)  d_in[2];
  const float* edge_feats = (const float*)d_in[3];
  // d_in[4] = timestamps (unused by reference)
  const float* W1   = (const float*)d_in[5];
  const float* b1   = (const float*)d_in[6];
  const float* W2   = (const float*)d_in[7];
  const float* b2   = (const float*)d_in[8];
  const float* W_ih = (const float*)d_in[9];
  const float* b_ih = (const float*)d_in[10];
  const float* W_hh = (const float*)d_in[11];
  const float* b_hh = (const float*)d_in[12];
  float* out = (float*)d_out;
  h16*   ws  = (h16*)d_ws;

  const int E = in_sizes[1];

  // 1) out = memory (non-updated rows)
  hipMemcpyAsync(d_out, (const void*)memory, (size_t)in_sizes[0] * sizeof(float),
                 hipMemcpyDeviceToDevice, stream);

  // 2) one-time weight prep: f16 conversions + algebraic fusion of the
  //    msg Linear2 into the GRU input weights:  gi = hid@(W_ih W2)^T + bf
  cvt_weights_kernel<<<192, 256, 0, stream>>>(W1, W_hh, ws);
  fuse_weights_kernel<<<192, 256, 0, stream>>>(W_ih, W2, b2, b_ih, ws);

  // 3) fused gather + MLP + GRU + scatter (8 waves/block, 32 edges/wave)
  const int blocks = (E + 255) / 256;
  tgn_update_kernel<<<blocks, 256, 0, stream>>>(memory, src_ids, dst_ids,
                                                edge_feats, ws, b1, b_hh, out, E);
}
